// CWVAE_25220047962790
// MI455X (gfx1250) — matrix operand order repaired and checked by reading, not compiled, not run on hardware
//
#include <hip/hip_runtime.h>

typedef __attribute__((ext_vector_type(16))) __bf16 v16bf;
typedef __attribute__((ext_vector_type(8)))  float  v8f;

union BfVec { uint4 q[2]; v16bf v; };

#define LEVELS 3
#define TAF    4
#define NB     32          /* batch */
#define T0_    64
#define STOCH_ 256
#define DETER_ 1024
#define EMB_   1024
#define OBSE_  1024

__device__ __forceinline__ unsigned short f2bf(float f) {
  unsigned u = __float_as_uint(f);
  unsigned r = u + 0x7fffu + ((u >> 16) & 1u);   // round-to-nearest-even
  return (unsigned short)(r >> 16);
}

// One wave computes the full M=32 x N=16 tile: two accumulators, B fragment
// loaded once per K-chunk and reused for both A fragments (halves W traffic).
// A: (32 x K) bf16 row-major.  W: (N x K) bf16 row-major, rows [0..15] of base.
__device__ __forceinline__ void bf_gemm_pair(v8f& c0, v8f& c1,
                                             const unsigned short* __restrict__ A, int lda,
                                             const unsigned short* __restrict__ W, int ldw,
                                             int K, int lane) {
  const int mr   = lane & 15;
  const int aoff = (lane >= 16) ? 8  : 0;   // A lanes >=16: K 8-15 / 24-31
  const int boff = (lane >= 16) ? 8  : 0;   // B lanes >=16: K 8-15 / 24-31
  const unsigned short* arow0 = A + (size_t)mr * lda;
  const unsigned short* arow1 = A + (size_t)(mr + 16) * lda;
  const unsigned short* wrow  = W + (size_t)mr * ldw;
#pragma unroll 2
  for (int k = 0; k < K; k += 32) {
    BfVec a0, a1, b;
    b.q[0]  = *(const uint4*)(wrow  + k + boff);
    b.q[1]  = *(const uint4*)(wrow  + k + boff + 16);
    a0.q[0] = *(const uint4*)(arow0 + k + aoff);
    a0.q[1] = *(const uint4*)(arow0 + k + aoff + 16);
    a1.q[0] = *(const uint4*)(arow1 + k + aoff);
    a1.q[1] = *(const uint4*)(arow1 + k + aoff + 16);
    c0 = __builtin_amdgcn_wmma_f32_16x16x32_bf16(false, a0.v, false, b.v,
                                                 (short)0, c0, false, false);
    c1 = __builtin_amdgcn_wmma_f32_16x16x32_bf16(false, a1.v, false, b.v,
                                                 (short)0, c1, false, false);
  }
}

__device__ __forceinline__ float apply_epi(float v, int epi) {
  if (epi == 1) return v > 0.f ? v : 0.f;
  if (epi == 2) {
    float x = v + 0.54f;
    return ((x > 30.f) ? x : log1pf(__expf(x))) + 1e-4f;
  }
  return v;
}

__device__ __forceinline__ void store_tile(const v8f& acc, int m_base, int col, float bv,
                                           int epi, float* __restrict__ outF, int ldF,
                                           unsigned short* __restrict__ outB, int ldB,
                                           int lane) {
  const int mofs = m_base + ((lane >= 16) ? 8 : 0);
#pragma unroll
  for (int j = 0; j < 8; ++j) {
    const int m = mofs + j;
    const float v = apply_epi(acc[j] + bv, epi);
    if (outF) outF[(size_t)m * ldF + col] = v;
    if (outB) outB[(size_t)m * ldB + col] = f2bf(v);
  }
}

// Generic GEMM: out[m][n] = epi( [A0|A1](m,:) . W(n,:) + bias[n] ), M = 32.
// Block = 32 threads (1 wave); blockIdx.x = N-tile.
__global__ __launch_bounds__(32) void gemm_bf16(
    const unsigned short* __restrict__ A0, int lda0, int K0,
    const unsigned short* __restrict__ A1, int lda1, int K1,
    const unsigned short* __restrict__ W,  int ldw,
    const float* __restrict__ bias,
    float* __restrict__ outF, int ldF,
    unsigned short* __restrict__ outB, int ldB,
    int epi)   // 0=linear 1=relu 2=softplus(x+0.54)+1e-4
{
  const int lane   = threadIdx.x & 31;
  const int n_base = blockIdx.x * 16;
  v8f c0 = {0.f,0.f,0.f,0.f,0.f,0.f,0.f,0.f};
  v8f c1 = c0;
  const unsigned short* wb = W + (size_t)n_base * ldw;
  bf_gemm_pair(c0, c1, A0, lda0, wb, ldw, K0, lane);
  if (K1 > 0) bf_gemm_pair(c0, c1, A1, lda1, wb + K0, ldw, K1, lane);

  const int col = n_base + (lane & 15);
  const float bv = bias ? bias[col] : 0.0f;
  store_tile(c0, 0,  col, bv, epi, outF, ldF, outB, ldB, lane);
  store_tile(c1, 16, col, bv, epi, outF, ldF, outB, ldB, lane);
}

// Two 256-wide heads sharing A: head1 linear (+optional bf16 copy), head2 softplus.
// Head2 skipped entirely when out2F == null (levels 1-2 need only qmean).
__global__ __launch_bounds__(32) void dual_head(
    const unsigned short* __restrict__ A, int K,
    const unsigned short* __restrict__ W1, const float* __restrict__ b1,
    float* __restrict__ out1F, unsigned short* __restrict__ out1B, int ldB1,
    const unsigned short* __restrict__ W2, const float* __restrict__ b2,
    float* __restrict__ out2F, int ldF)
{
  const int lane   = threadIdx.x & 31;
  const int n_base = blockIdx.x * 16;
  const int col    = n_base + (lane & 15);
  const v8f z8 = {0.f,0.f,0.f,0.f,0.f,0.f,0.f,0.f};

  v8f c0 = z8, c1 = z8;
  bf_gemm_pair(c0, c1, A, K, W1 + (size_t)n_base * K, K, K, lane);
  const float bv1 = b1[col];
  store_tile(c0, 0,  col, bv1, 0, out1F, ldF, out1B, ldB1, lane);
  store_tile(c1, 16, col, bv1, 0, out1F, ldF, out1B, ldB1, lane);

  if (out2F) {
    v8f d0 = z8, d1 = z8;
    bf_gemm_pair(d0, d1, A, K, W2 + (size_t)n_base * K, K, K, lane);
    const float bv2 = b2[col];
    store_tile(d0, 0,  col, bv2, 2, out2F, ldF, nullptr, 0, lane);
    store_tile(d1, 16, col, bv2, 2, out2F, ldF, nullptr, 0, lane);
  }
}

// Full GRU step for a 32x16 det tile: 6 K=1024 WMMA pair-accumulations + gate combine.
__global__ __launch_bounds__(32) void gru_step(
    const unsigned short* __restrict__ h1,     // (32,1024) bf16
    const unsigned short* __restrict__ detB,   // h(t-1) bf16
    const float*          __restrict__ detF,   // h(t-1) fp32
    const unsigned short* __restrict__ wih,    // (3072,1024) bf16
    const unsigned short* __restrict__ whh,    // (3072,1024) bf16
    const float* __restrict__ bih, const float* __restrict__ bhh,
    float* __restrict__ detF_out, unsigned short* __restrict__ detB_out,
    unsigned short* __restrict__ seq_slot,     // (32,1024) bf16 or null
    float* __restrict__ out0, int ld_out)      // level-0 output or null
{
  const int lane   = threadIdx.x & 31;
  const int n_base = blockIdx.x * 16;
  const v8f z8 = {0.f,0.f,0.f,0.f,0.f,0.f,0.f,0.f};

  v8f ai0[3], ai1[3], ah0[3], ah1[3];
#pragma unroll
  for (int g = 0; g < 3; ++g) {
    const size_t wrow = (size_t)(g * DETER_ + n_base) * DETER_;
    ai0[g] = z8; ai1[g] = z8; ah0[g] = z8; ah1[g] = z8;
    bf_gemm_pair(ai0[g], ai1[g], h1,   DETER_, wih + wrow, DETER_, DETER_, lane);
    bf_gemm_pair(ah0[g], ah1[g], detB, DETER_, whh + wrow, DETER_, DETER_, lane);
  }

  const int col  = n_base + (lane & 15);
  const int mofs = (lane >= 16) ? 8 : 0;
  const float br = bih[col],            cr = bhh[col];
  const float bz = bih[col + DETER_],   cz = bhh[col + DETER_];
  const float bn = bih[col + 2*DETER_], cn = bhh[col + 2*DETER_];
#pragma unroll
  for (int half = 0; half < 2; ++half) {
    const v8f& xr = half ? ai1[0] : ai0[0];  const v8f& hr = half ? ah1[0] : ah0[0];
    const v8f& xz = half ? ai1[1] : ai0[1];  const v8f& hz = half ? ah1[1] : ah0[1];
    const v8f& xn = half ? ai1[2] : ai0[2];  const v8f& hn = half ? ah1[2] : ah0[2];
#pragma unroll
    for (int j = 0; j < 8; ++j) {
      const int m = half * 16 + mofs + j;
      const float r  = 1.f / (1.f + __expf(-(xr[j] + br + hr[j] + cr)));
      const float z  = 1.f / (1.f + __expf(-(xz[j] + bz + hz[j] + cz)));
      const float nn = tanhf(xn[j] + bn + r * (hn[j] + cn));
      const float h  = detF[(size_t)m * DETER_ + col];
      const float d  = (1.f - z) * nn + z * h;
      detF_out[(size_t)m * DETER_ + col] = d;
      const unsigned short db = f2bf(d);
      detB_out[(size_t)m * DETER_ + col] = db;
      if (seq_slot) seq_slot[(size_t)m * DETER_ + col] = db;
      if (out0)     out0[(size_t)m * ld_out + col] = d;
    }
  }
}

__global__ void f32_to_bf16_kernel(const float* __restrict__ src,
                                   unsigned short* __restrict__ dst, long n) {
  long i = (long)blockIdx.x * blockDim.x + threadIdx.x;
  const long stride = (long)gridDim.x * blockDim.x;
  for (; i < n; i += stride) dst[i] = f2bf(src[i]);
}

__global__ void zero_u32_kernel(unsigned int* __restrict__ p, long n) {
  long i = (long)blockIdx.x * blockDim.x + threadIdx.x;
  const long stride = (long)gridDim.x * blockDim.x;
  for (; i < n; i += stride) p[i] = 0u;
}

static inline void cvt(const float* s, unsigned short* d, long n, hipStream_t st) {
  int blocks = (int)((n + 255) / 256); if (blocks > 4096) blocks = 4096;
  f32_to_bf16_kernel<<<blocks, 256, 0, st>>>(s, d, n);
}
static inline void zero(void* p, size_t bytes, hipStream_t st) {
  long n = (long)(bytes / 4);
  int blocks = (int)((n + 255) / 256); if (blocks > 2048) blocks = 2048;
  zero_u32_kernel<<<blocks, 256, 0, st>>>((unsigned int*)p, n);
}

extern "C" void kernel_launch(void* const* d_in, const int* in_sizes, int n_in,
                              void* d_out, int out_size, void* d_ws, size_t ws_size,
                              hipStream_t stream) {
  const float* obsF[3] = {(const float*)d_in[0], (const float*)d_in[1], (const float*)d_in[2]};
  const float* ph1_w = (const float*)d_in[3];  const float* ph1_b = (const float*)d_in[4];
  const float* wih_f = (const float*)d_in[5];  const float* whh_f = (const float*)d_in[6];
  const float* bih   = (const float*)d_in[7];  const float* bhh   = (const float*)d_in[8];
  const float* pm_w  = (const float*)d_in[9];  const float* pm_b  = (const float*)d_in[10];
  const float* ps_w  = (const float*)d_in[11]; const float* ps_b  = (const float*)d_in[12];
  const float* qh1_w = (const float*)d_in[13]; const float* qh1_b = (const float*)d_in[14];
  const float* qm_w  = (const float*)d_in[15]; const float* qm_b  = (const float*)d_in[16];
  const float* qs_w  = (const float*)d_in[17]; const float* qs_b  = (const float*)d_in[18];
  float* out = (float*)d_out;

  // ---- workspace layout (256B aligned slabs) ----
  char* ws = (char*)d_ws;
  size_t off = 0;
  auto alloc = [&](size_t bytes) -> char* {
    char* p = ws + off;
    off = (off + bytes + 255) & ~(size_t)255;
    return p;
  };
  const size_t sPH1 = (size_t)LEVELS * EMB_ * (STOCH_ + DETER_);
  const size_t sGRU = (size_t)LEVELS * 3 * DETER_ * EMB_;
  const size_t sHEAD= (size_t)LEVELS * STOCH_ * DETER_;
  const size_t sQH1 = (size_t)LEVELS * EMB_ * (DETER_ + OBSE_);
  unsigned short* w_ph1 = (unsigned short*)alloc(sPH1 * 2);
  unsigned short* w_wih = (unsigned short*)alloc(sGRU * 2);
  unsigned short* w_whh = (unsigned short*)alloc(sGRU * 2);
  unsigned short* w_pm  = (unsigned short*)alloc(sHEAD * 2);
  unsigned short* w_ps  = (unsigned short*)alloc(sHEAD * 2);
  unsigned short* w_qm  = (unsigned short*)alloc(sHEAD * 2);
  unsigned short* w_qs  = (unsigned short*)alloc(sHEAD * 2);
  unsigned short* w_qh1 = (unsigned short*)alloc(sQH1 * 2);
  const int Ts[3] = {T0_, T0_ / TAF, T0_ / TAF / TAF};
  unsigned short* obsB[3];
  for (int l = 0; l < 3; ++l) obsB[l] = (unsigned short*)alloc((size_t)NB * Ts[l] * OBSE_ * 2);
  unsigned short* seqB[3] = {nullptr, nullptr, nullptr};
  seqB[1] = (unsigned short*)alloc((size_t)Ts[1] * NB * DETER_ * 2);
  seqB[2] = (unsigned short*)alloc((size_t)Ts[2] * NB * DETER_ * 2);
  unsigned short* zctx = (unsigned short*)alloc((size_t)NB * DETER_ * 2);
  // state region (zeroed per level in one shot)
  size_t state_beg = off;
  unsigned short* stoch  = (unsigned short*)alloc((size_t)NB * STOCH_ * 2);
  float*          detF[2]; unsigned short* detB[2];
  detF[0] = (float*)alloc((size_t)NB * DETER_ * 4);
  detF[1] = (float*)alloc((size_t)NB * DETER_ * 4);
  detB[0] = (unsigned short*)alloc((size_t)NB * DETER_ * 2);
  detB[1] = (unsigned short*)alloc((size_t)NB * DETER_ * 2);
  size_t state_end = off;
  unsigned short* h1 = (unsigned short*)alloc((size_t)NB * EMB_ * 2);
  unsigned short* qh = (unsigned short*)alloc((size_t)NB * EMB_ * 2);

  // ---- one-time (per call) conversions ----
  cvt(ph1_w, w_ph1, (long)sPH1, stream);
  cvt(wih_f, w_wih, (long)sGRU, stream);
  cvt(whh_f, w_whh, (long)sGRU, stream);
  cvt(pm_w,  w_pm,  (long)sHEAD, stream);
  cvt(ps_w,  w_ps,  (long)sHEAD, stream);
  cvt(qm_w,  w_qm,  (long)sHEAD, stream);
  cvt(qs_w,  w_qs,  (long)sHEAD, stream);
  cvt(qh1_w, w_qh1, (long)sQH1, stream);
  for (int l = 0; l < 3; ++l) cvt(obsF[l], obsB[l], (long)NB * Ts[l] * OBSE_, stream);
  zero(zctx, (size_t)NB * DETER_ * 2, stream);

  const int ldOut = T0_ * 2048;
  const dim3 blk(32);

  for (int lv = 2; lv >= 0; --lv) {
    const int T = Ts[lv];
    zero(ws + state_beg, state_end - state_beg, stream);   // stoch0 = det0 = 0
    int cur = 0;
    const size_t wofs_ph1 = (size_t)lv * EMB_ * (STOCH_ + DETER_);
    const size_t wofs_gru = (size_t)lv * 3 * DETER_ * EMB_;
    const size_t wofs_hd  = (size_t)lv * STOCH_ * DETER_;
    const size_t wofs_qh1 = (size_t)lv * EMB_ * (DETER_ + OBSE_);
    for (int t = 0; t < T; ++t) {
      const int nx = cur ^ 1;
      const unsigned short* ctx =
          (lv == 2) ? zctx : seqB[lv + 1] + (size_t)(t / TAF) * NB * DETER_;
      // h1 = relu([stoch | ctx] @ ph1_w^T + b)
      gemm_bf16<<<dim3(EMB_ / 16), blk, 0, stream>>>(
          stoch, STOCH_, STOCH_, ctx, DETER_, DETER_,
          w_ph1 + wofs_ph1, STOCH_ + DETER_, ph1_b + (size_t)lv * EMB_,
          nullptr, 0, h1, EMB_, 1);
      // det_new = GRU(h1, det)
      gru_step<<<dim3(DETER_ / 16), blk, 0, stream>>>(
          h1, detB[cur], detF[cur],
          w_wih + wofs_gru, w_whh + wofs_gru,
          bih + (size_t)lv * 3 * DETER_, bhh + (size_t)lv * 3 * DETER_,
          detF[nx], detB[nx],
          (lv > 0) ? seqB[lv] + (size_t)t * NB * DETER_ : nullptr,
          (lv == 0) ? out + (size_t)t * 2048 : nullptr, ldOut);
      if (lv == 0) {
        // pmean (linear) + pstd (softplus) from det_new
        dual_head<<<dim3(STOCH_ / 16), blk, 0, stream>>>(
            detB[nx], DETER_,
            w_pm + wofs_hd, pm_b + (size_t)lv * STOCH_,
            out + (size_t)t * 2048 + 1024, nullptr, 0,
            w_ps + wofs_hd, ps_b + (size_t)lv * STOCH_,
            out + (size_t)t * 2048 + 1280, ldOut);
      }
      // qh = relu([det_new | obs_t] @ qh1_w^T + b)
      gemm_bf16<<<dim3(EMB_ / 16), blk, 0, stream>>>(
          detB[nx], DETER_, DETER_, obsB[lv] + (size_t)t * OBSE_, T * OBSE_, OBSE_,
          w_qh1 + wofs_qh1, DETER_ + OBSE_, qh1_b + (size_t)lv * EMB_,
          nullptr, 0, qh, EMB_, 1);
      // qmean (linear -> stoch, + out at lv0) + qstd (softplus, lv0 only)
      dual_head<<<dim3(STOCH_ / 16), blk, 0, stream>>>(
          qh, EMB_,
          w_qm + wofs_hd, qm_b + (size_t)lv * STOCH_,
          (lv == 0) ? out + (size_t)t * 2048 + 1536 : nullptr, stoch, STOCH_,
          w_qs + wofs_hd, qs_b + (size_t)lv * STOCH_,
          (lv == 0) ? out + (size_t)t * 2048 + 1792 : nullptr, ldOut);
      cur = nx;
    }
  }
}
